// MixtureOfExperts_12601434047133
// MI455X (gfx1250) — matrix operand order repaired and checked by reading, not compiled, not run on hardware
//
#include <hip/hip_runtime.h>

// ---------------------------------------------------------------------------
// MoE forward for MI455X (gfx1250, wave32, WMMA).
// Compute-bound (~600 FLOP/byte) -> bf16 WMMA path (v_wmma_f32_16x16x32_bf16),
// f32 accumulate. Weights converted once per call to bf16 scratch; per-expert
// bf16 weights (~35 MB) stay L2-resident (192 MB) during that expert's tiles.
// GEMM staging uses global_load_async_to_lds_b128 (ASYNCcnt) with ping-pong
// LDS buffers when available; otherwise register-double-buffered loads.
// ---------------------------------------------------------------------------

typedef unsigned short u16;
typedef __attribute__((ext_vector_type(16))) __bf16 v16bf;
typedef __attribute__((ext_vector_type(8)))  float  v8f;
typedef int v4i __attribute__((vector_size(16)));   // matches builtin param type

#define DEVFN __device__ __forceinline__
#define AS1 __attribute__((address_space(1)))
#define AS3 __attribute__((address_space(3)))

#if __has_builtin(__builtin_amdgcn_global_load_async_to_lds_b128)
#define ASYNC_LDS 1
#else
#define ASYNC_LDS 0
#endif

#if __has_builtin(__builtin_amdgcn_s_wait_asynccnt)
#define WAIT_ASYNC(n) __builtin_amdgcn_s_wait_asynccnt(n)
#else
#define WAIT_ASYNC(n) asm volatile("s_wait_asynccnt %0" ::"i"(n) : "memory")
#endif

constexpr int NBUF = ASYNC_LDS ? 2 : 1;   // LDS ping-pong only on async path

// Problem constants (from reference setup_inputs)
constexpr int Bc = 2, Tc = 4096, Dc = 1024, Ec = 8, Hc = 5632;
constexpr int Nc  = Bc * Tc;          // 8192 tokens
constexpr int H2c = 2 * Hc;           // 11264
constexpr int CAPc = 1280;            // ceil(N/E * 1.25)

// d_out layout (floats): output, aux_loss, top, keep  (concatenated flat)
constexpr size_t OUT_AUX  = (size_t)Nc * Dc;
constexpr size_t OUT_TOP  = OUT_AUX + 1;
constexpr size_t OUT_KEEP = OUT_TOP + Nc;

// Workspace layout (bytes)
constexpr size_t WS_ACC  = 0;                                  // sp[8]@0 cnt[8]@64 z2@128
constexpr size_t WS_TOP  = 256;
constexpr size_t WS_SLOT = WS_TOP  + sizeof(int) * (size_t)Nc;
constexpr size_t WS_KEEP = WS_SLOT + sizeof(int) * (size_t)Nc;
constexpr size_t WS_XE   = WS_KEEP + sizeof(int) * (size_t)Nc;           // bf16 E*C*D
constexpr size_t WS_W13  = WS_XE   + 2ull * Ec * CAPc * Dc;              // bf16 E*2H*D
constexpr size_t WS_W2   = WS_W13  + 2ull * Ec * H2c * Dc;               // bf16 E*D*H
constexpr size_t WS_H    = WS_W2   + 2ull * Ec * Dc * Hc;                // bf16 E*C*H
constexpr size_t WS_YE   = WS_H    + 2ull * Ec * CAPc * Hc;              // f32  E*C*D

DEVFN u16 f32_to_bf16(float f) {
  unsigned int u = __float_as_uint(f);
  unsigned int r = u + 0x7FFFu + ((u >> 16) & 1u);   // round-to-nearest-even
  return (u16)(r >> 16);
}

#if ASYNC_LDS
DEVFN void async_copy_b128(const u16* gp, const u16* lp) {
  __builtin_amdgcn_global_load_async_to_lds_b128(
      (AS1 v4i*)(uintptr_t)gp,
      (AS3 v4i*)(unsigned)(uintptr_t)lp, 0, 0);
}
#endif

// A operand, 16x32 bf16 (ISA 7.12.2 "16-bit A-Matrix 16x32"):
//   lanes 0-15 : row = lane,    K = {0..7} then {16..23}
//   lanes 16-31: row = lane-16, K = {8..15} then {24..31}
DEVFN v16bf load_fragA(const u16* rowbase, int lane) {
  const u16* p = rowbase + (size_t)(lane & 15) * 40 + ((lane >> 4) << 3);
  struct U32x8 { uint4 a, b; } u;
  u.a = *(const uint4*)(p);
  u.b = *(const uint4*)(p + 16);
  return __builtin_bit_cast(v16bf, u);
}

// B operand, 32x16 bf16 (same K order as A):
//   lane = column (lane&15); lanes 0-15 hold K={0..7} then {16..23},
//   lanes 16-31 hold K={8..15} then {24..31}.
DEVFN v16bf load_fragB(const u16* rowbase, int lane) {
  const u16* p = rowbase + (size_t)(lane & 15) * 40 + ((lane >> 4) << 3);
  struct U32x8 { uint4 a, b; } u;
  u.a = *(const uint4*)(p);
  u.b = *(const uint4*)(p + 16);
  return __builtin_bit_cast(v16bf, u);
}

DEVFN v8f wmma_bf16(v16bf a, v16bf b, v8f c) {
  return __builtin_amdgcn_wmma_f32_16x16x32_bf16(
      false, a, false, b, (short)0, c, false, false);
}

// ---------------------------------------------------------------------------
__global__ void zero_acc_kernel(float* sp, int* cnt, float* z2) {
  int t = threadIdx.x;
  if (t < Ec) { sp[t] = 0.0f; cnt[t] = 0; }
  if (t == 0) *z2 = 0.0f;
}

__global__ void convert_bf16_kernel(const float* __restrict__ src,
                                    u16* __restrict__ dst, size_t n) {
  size_t i = (size_t)blockIdx.x * blockDim.x + threadIdx.x;
  size_t stride = (size_t)gridDim.x * blockDim.x;
  for (; i < n; i += stride) dst[i] = f32_to_bf16(src[i]);
}

// Router: one wave per token; wave32 shuffle reduction.
__global__ __launch_bounds__(256)
void router_kernel(const float* __restrict__ x, const float* __restrict__ wg,
                   int* __restrict__ top, float* sp, int* cnt, float* z2acc) {
  const int lane = threadIdx.x & 31, wave = threadIdx.x >> 5;
  const int i = blockIdx.x * 8 + wave;
  float acc[Ec];
#pragma unroll
  for (int e = 0; e < Ec; ++e) acc[e] = 0.0f;
  const float* xr = x + (size_t)i * Dc;
  for (int d = lane; d < Dc; d += 32) {
    float xv = xr[d];
#pragma unroll
    for (int e = 0; e < Ec; ++e) acc[e] += xv * wg[e * Dc + d];
  }
#pragma unroll
  for (int e = 0; e < Ec; ++e)
    for (int off = 16; off; off >>= 1) acc[e] += __shfl_xor(acc[e], off, 32);
  if (lane == 0) {
    float m = acc[0]; int t = 0;
#pragma unroll
    for (int e = 1; e < Ec; ++e) if (acc[e] > m) { m = acc[e]; t = e; }
    float p[Ec], s = 0.0f;
#pragma unroll
    for (int e = 0; e < Ec; ++e) { p[e] = __expf(acc[e] - m); s += p[e]; }
    float inv = 1.0f / s;
#pragma unroll
    for (int e = 0; e < Ec; ++e) atomicAdd(&sp[e], p[e] * inv);
    atomicAdd(&cnt[t], 1);
    float z = m + __logf(s);
    atomicAdd(z2acc, z * z);
    top[i] = t;
  }
}

// Ordered capacity scan: single block, ballot-based per-expert rank so token
// order matches the reference cumsum exactly.
__global__ __launch_bounds__(256)
void scan_kernel(const int* __restrict__ top, int* __restrict__ slot,
                 int* __restrict__ keep) {
  __shared__ int wcnt[8][Ec];
  __shared__ int base[Ec];
  const int tid = threadIdx.x, lane = tid & 31, wave = tid >> 5;
  if (tid < Ec) base[tid] = 0;
  __syncthreads();
  for (int start = 0; start < Nc; start += 256) {
    const int i = start + tid;
    const int e = top[i];
    int rank = 0;
#pragma unroll
    for (int ex = 0; ex < Ec; ++ex) {
      unsigned long long m = __ballot(e == ex);
      if (lane == 0) wcnt[wave][ex] = __popcll(m);
      if (e == ex) rank = __popcll(m & ((1ull << lane) - 1ull));
    }
    __syncthreads();
    int before = base[e];
    for (int w = 0; w < wave; ++w) before += wcnt[w][e];
    const int pos = before + rank;
    const int kp = pos < CAPc;
    slot[i] = kp ? pos : (CAPc - 1);
    keep[i] = kp;
    __syncthreads();
    if (tid < Ec) {
      int tot = 0;
      for (int w = 0; w < 8; ++w) tot += wcnt[w][tid];
      base[tid] += tot;
    }
    __syncthreads();
  }
}

__global__ __launch_bounds__(256)
void scatter_kernel(const float* __restrict__ x, const int* __restrict__ top,
                    const int* __restrict__ slot, const int* __restrict__ keep,
                    u16* __restrict__ xe) {
  const int i = blockIdx.x;
  if (!keep[i]) return;
  const float* src = x + (size_t)i * Dc;
  u16* dst = xe + (size_t)(top[i] * CAPc + slot[i]) * Dc;
  for (int c = threadIdx.x; c < Dc; c += 256) dst[c] = f32_to_bf16(src[c]);
}

// GEMM1 + fused SiLU*u. Block tile 128(M) x 64(N of h); gate rows [n0,n0+64)
// and up rows [H+n0,H+n0+64) both staged so silu(g)*u fuses in the epilogue.
// 8 waves: 2(M) x 4(N). Per thread per tile: 4 x b128 transfers.
__global__ __launch_bounds__(256)
void gemm1_silu_kernel(const u16* __restrict__ xe, const u16* __restrict__ w13,
                       u16* __restrict__ h) {
  __shared__ __align__(16) u16 As[NBUF][128][40];
  __shared__ __align__(16) u16 Bg[NBUF][64][40];
  __shared__ __align__(16) u16 Bu[NBUF][64][40];
  const int e = blockIdx.z, m0 = blockIdx.y * 128, n0 = blockIdx.x * 64;
  const int tid = threadIdx.x, lane = tid & 31, wave = tid >> 5;
  const int wm = wave & 1, wn = wave >> 1;
  const int r0 = tid >> 2, seg = tid & 3;       // r0: 0..63
  v8f accg[4], accu[4];
#pragma unroll
  for (int i = 0; i < 4; ++i) { accg[i] = {}; accu[i] = {}; }

  const u16* gA0 = xe  + (size_t)(e * CAPc + m0 + r0) * Dc + seg * 8;
  const u16* gA1 = xe  + (size_t)(e * CAPc + m0 + r0 + 64) * Dc + seg * 8;
  const u16* gBg = w13 + ((size_t)e * H2c + n0 + r0) * Dc + seg * 8;
  const u16* gBu = w13 + ((size_t)e * H2c + Hc + n0 + r0) * Dc + seg * 8;

  auto compute = [&](int buf) {
    const v16bf bg = load_fragB(&Bg[buf][wn * 16][0], lane);
    const v16bf bu = load_fragB(&Bu[buf][wn * 16][0], lane);
#pragma unroll
    for (int i = 0; i < 4; ++i) {
      const v16bf a = load_fragA(&As[buf][wm * 64 + i * 16][0], lane);
      accg[i] = wmma_bf16(a, bg, accg[i]);
      accu[i] = wmma_bf16(a, bu, accu[i]);
    }
  };

#if ASYNC_LDS
  auto issue = [&](int buf, int k0) {
    async_copy_b128(gA0 + k0, &As[buf][r0][seg * 8]);
    async_copy_b128(gA1 + k0, &As[buf][r0 + 64][seg * 8]);
    async_copy_b128(gBg + k0, &Bg[buf][r0][seg * 8]);
    async_copy_b128(gBu + k0, &Bu[buf][r0][seg * 8]);
  };
  constexpr int NK = Dc / 32;
  issue(0, 0);
  for (int it = 0; it < NK; ++it) {
    const int cur = it & 1;
    if (it) __syncthreads();                  // buf cur^1 consumers done
    if (it + 1 < NK) { issue(cur ^ 1, (it + 1) * 32); WAIT_ASYNC(4); }
    else             { WAIT_ASYNC(0); }
    __syncthreads();                          // all waves' tile landed
    compute(cur);
  }
#else
  uint4 rA0, rA1, rBg, rBu;
  auto fetch = [&](int k0) {
    rA0 = *(const uint4*)(gA0 + k0);
    rA1 = *(const uint4*)(gA1 + k0);
    rBg = *(const uint4*)(gBg + k0);
    rBu = *(const uint4*)(gBu + k0);
  };
  fetch(0);
  for (int k0 = 0; k0 < Dc; k0 += 32) {
    __syncthreads();
    *(uint4*)&As[0][r0][seg * 8] = rA0;
    *(uint4*)&As[0][r0 + 64][seg * 8] = rA1;
    *(uint4*)&Bg[0][r0][seg * 8] = rBg;
    *(uint4*)&Bu[0][r0][seg * 8] = rBu;
    __syncthreads();
    if (k0 + 32 < Dc) { fetch(k0 + 32); __builtin_prefetch(gA0 + k0 + 64, 0, 1); }
    compute(0);
  }
#endif

  // Epilogue: h = silu(g) * u, bf16 store.
  const int col = n0 + wn * 16 + (lane & 15);
  const int rowHalf = (lane >> 4) * 8;
#pragma unroll
  for (int i = 0; i < 4; ++i)
#pragma unroll
    for (int v = 0; v < 8; ++v) {
      const int row = m0 + wm * 64 + i * 16 + v + rowHalf;
      const float g = accg[i][v], u = accu[i][v];
      const float hv = (g / (1.0f + __expf(-g))) * u;
      h[(size_t)(e * CAPc + row) * Hc + col] = f32_to_bf16(hv);
    }
}

// GEMM2: ye[e] = h[e] (C x H) x W2[e]^T (D x H), contraction over H.
// Block tile 128x128. 8 waves: 2(M) x 4(N, 32 cols).
__global__ __launch_bounds__(256)
void gemm2_kernel(const u16* __restrict__ h, const u16* __restrict__ w2,
                  float* __restrict__ ye) {
  __shared__ __align__(16) u16 As[NBUF][128][40];
  __shared__ __align__(16) u16 Bs[NBUF][128][40];
  const int e = blockIdx.z, m0 = blockIdx.y * 128, n0 = blockIdx.x * 128;
  const int tid = threadIdx.x, lane = tid & 31, wave = tid >> 5;
  const int wm = wave & 1, wn = wave >> 1;
  const int r0 = tid >> 2, seg = tid & 3;
  v8f acc[4][2];
#pragma unroll
  for (int i = 0; i < 4; ++i) { acc[i][0] = {}; acc[i][1] = {}; }

  const u16* gA0 = h  + (size_t)(e * CAPc + m0 + r0) * Hc + seg * 8;
  const u16* gA1 = h  + (size_t)(e * CAPc + m0 + r0 + 64) * Hc + seg * 8;
  const u16* gB0 = w2 + (size_t)(e * Dc + n0 + r0) * Hc + seg * 8;
  const u16* gB1 = w2 + (size_t)(e * Dc + n0 + r0 + 64) * Hc + seg * 8;

  auto compute = [&](int buf) {
    const v16bf b0 = load_fragB(&Bs[buf][wn * 32][0], lane);
    const v16bf b1 = load_fragB(&Bs[buf][wn * 32 + 16][0], lane);
#pragma unroll
    for (int i = 0; i < 4; ++i) {
      const v16bf a = load_fragA(&As[buf][wm * 64 + i * 16][0], lane);
      acc[i][0] = wmma_bf16(a, b0, acc[i][0]);
      acc[i][1] = wmma_bf16(a, b1, acc[i][1]);
    }
  };

#if ASYNC_LDS
  auto issue = [&](int buf, int k0) {
    async_copy_b128(gA0 + k0, &As[buf][r0][seg * 8]);
    async_copy_b128(gA1 + k0, &As[buf][r0 + 64][seg * 8]);
    async_copy_b128(gB0 + k0, &Bs[buf][r0][seg * 8]);
    async_copy_b128(gB1 + k0, &Bs[buf][r0 + 64][seg * 8]);
  };
  constexpr int NK = Hc / 32;
  issue(0, 0);
  for (int it = 0; it < NK; ++it) {
    const int cur = it & 1;
    if (it) __syncthreads();
    if (it + 1 < NK) { issue(cur ^ 1, (it + 1) * 32); WAIT_ASYNC(4); }
    else             { WAIT_ASYNC(0); }
    __syncthreads();
    compute(cur);
  }
#else
  uint4 rA0, rA1, rB0, rB1;
  auto fetch = [&](int k0) {
    rA0 = *(const uint4*)(gA0 + k0);
    rA1 = *(const uint4*)(gA1 + k0);
    rB0 = *(const uint4*)(gB0 + k0);
    rB1 = *(const uint4*)(gB1 + k0);
  };
  fetch(0);
  for (int k0 = 0; k0 < Hc; k0 += 32) {
    __syncthreads();
    *(uint4*)&As[0][r0][seg * 8] = rA0;
    *(uint4*)&As[0][r0 + 64][seg * 8] = rA1;
    *(uint4*)&Bs[0][r0][seg * 8] = rB0;
    *(uint4*)&Bs[0][r0 + 64][seg * 8] = rB1;
    __syncthreads();
    if (k0 + 32 < Hc) { fetch(k0 + 32); __builtin_prefetch(gA0 + k0 + 64, 0, 1); }
    compute(0);
  }
#endif

  const int c0 = lane & 15, rowHalf = (lane >> 4) * 8;
#pragma unroll
  for (int i = 0; i < 4; ++i)
#pragma unroll
    for (int j = 0; j < 2; ++j)
#pragma unroll
      for (int v = 0; v < 8; ++v) {
        const int row = m0 + wm * 64 + i * 16 + v + rowHalf;
        const int col = n0 + wn * 32 + j * 16 + c0;
        ye[(size_t)(e * CAPc + row) * Dc + col] = acc[i][j][v];
      }
}

// Gather expert outputs back to token order; emit aux_loss / top / keep.
__global__ __launch_bounds__(256)
void gather_kernel(const float* __restrict__ ye, const int* __restrict__ top,
                   const int* __restrict__ slot, const int* __restrict__ keep,
                   const float* __restrict__ sp, const int* __restrict__ cnt,
                   const float* __restrict__ z2, float* __restrict__ out) {
  const int i = blockIdx.x;
  const int kp = keep[i];
  float* dst = out + (size_t)i * Dc;
  if (kp) {
    const float* src = ye + (size_t)(top[i] * CAPc + slot[i]) * Dc;
    for (int c = threadIdx.x; c < Dc; c += 256) dst[c] = src[c];
  } else {
    for (int c = threadIdx.x; c < Dc; c += 256) dst[c] = 0.0f;
  }
  if (threadIdx.x == 0) {
    out[OUT_TOP + i]  = (float)top[i];
    out[OUT_KEEP + i] = kp ? 1.0f : 0.0f;
  }
  if (i == 0 && threadIdx.x == 1) {
    float bal = 0.0f;
#pragma unroll
    for (int e = 0; e < Ec; ++e)
      bal += (sp[e] / (float)Nc) * ((float)cnt[e] / (float)Nc);
    out[OUT_AUX] = bal * 0.01f * (float)Ec + ((*z2) / (float)Nc) * 0.001f;
  }
}

// ---------------------------------------------------------------------------
extern "C" void kernel_launch(void* const* d_in, const int* in_sizes, int n_in,
                              void* d_out, int out_size, void* d_ws, size_t ws_size,
                              hipStream_t stream) {
  (void)in_sizes; (void)n_in; (void)out_size; (void)ws_size;
  const float* x   = (const float*)d_in[0];
  const float* wg  = (const float*)d_in[1];
  const float* w13 = (const float*)d_in[2];
  const float* w2  = (const float*)d_in[3];
  float* out = (float*)d_out;
  char* ws = (char*)d_ws;

  float* sp    = (float*)(ws + WS_ACC);
  int*   cnt   = (int*)  (ws + WS_ACC + 64);
  float* z2    = (float*)(ws + WS_ACC + 128);
  int*   top   = (int*)  (ws + WS_TOP);
  int*   slot  = (int*)  (ws + WS_SLOT);
  int*   keep  = (int*)  (ws + WS_KEEP);
  u16*   xe    = (u16*)  (ws + WS_XE);
  u16*   w13b  = (u16*)  (ws + WS_W13);
  u16*   w2b   = (u16*)  (ws + WS_W2);
  u16*   hbuf  = (u16*)  (ws + WS_H);
  float* ye    = (float*)(ws + WS_YE);

  zero_acc_kernel<<<1, 64, 0, stream>>>(sp, cnt, z2);
  convert_bf16_kernel<<<4096, 256, 0, stream>>>(w13, w13b, (size_t)Ec * H2c * Dc);
  convert_bf16_kernel<<<4096, 256, 0, stream>>>(w2,  w2b,  (size_t)Ec * Dc * Hc);
  router_kernel<<<Nc / 8, 256, 0, stream>>>(x, wg, top, sp, cnt, z2);
  scan_kernel<<<1, 256, 0, stream>>>(top, slot, keep);
  scatter_kernel<<<Nc, 256, 0, stream>>>(x, top, slot, keep, xe);
  gemm1_silu_kernel<<<dim3(Hc / 64, CAPc / 128, Ec), 256, 0, stream>>>(xe, w13b, hbuf);
  gemm2_kernel<<<dim3(Dc / 128, CAPc / 128, Ec), 256, 0, stream>>>(hbuf, w2b, ye);
  gather_kernel<<<Nc, 256, 0, stream>>>(ye, top, slot, keep, sp, cnt, z2, out);
}
